// Decoder_71210557768030
// MI455X (gfx1250) — matrix operand order repaired and checked by reading, not compiled, not run on hardware
//
#include <hip/hip_runtime.h>
#include <math.h>
#include <stdint.h>

// ---------------- problem constants ----------------
constexpr int Vv = 32000;
constexpr int Dm = 1024;
constexpr int Hh = 16;
constexpr int Ll = 8;
constexpr int Bb = 2;
constexpr int Tt = 1024;
constexpr int BT = Bb * Tt;      // 2048
constexpr int HD = Dm / Hh;      // 64
constexpr float LN_EPS = 1e-5f;

// ---------------- WMMA / bf16 types ----------------
typedef __attribute__((ext_vector_type(16))) __bf16 bf16x16;
typedef __attribute__((ext_vector_type(2)))  __bf16 bf16x2;
typedef __attribute__((ext_vector_type(2)))  float  f32x2;
typedef __attribute__((ext_vector_type(8)))  float  v8f;

union Frag {
    bf16x16  v;
    unsigned u[8];
};

// native bf16 converts (v_cvt_pk_bf16_f32 on gfx1250)
__device__ inline unsigned pack2(float a, float b) {
    f32x2 f = {a, b};
    union { bf16x2 v; unsigned u; } x;
    x.v = __builtin_convertvector(f, bf16x2);
    return x.u;
}
__device__ inline unsigned short f2bf(float f) {
    union { __bf16 h; unsigned short s; } x;
    x.h = (__bf16)f;
    return x.s;
}
__device__ inline v8f wmma_bf16(const Frag& a, const Frag& b, v8f c) {
    return __builtin_amdgcn_wmma_f32_16x16x32_bf16(
        false, a.v, false, b.v, (short)0, c, false, false);
}
__device__ inline float wave_sum(float v) {
    for (int o = 16; o > 0; o >>= 1) v += __shfl_xor(v, o, 32);
    return v;
}
__device__ inline float wave_max(float v) {
    for (int o = 16; o > 0; o >>= 1) v = fmaxf(v, __shfl_xor(v, o, 32));
    return v;
}
// load 16B-aligned run of 4 uints into fragment half
__device__ inline void frag_lo(Frag& f, const void* p) {
    uint4 t = *(const uint4*)p;
    f.u[0] = t.x; f.u[1] = t.y; f.u[2] = t.z; f.u[3] = t.w;
}
__device__ inline void frag_hi(Frag& f, const void* p) {
    uint4 t = *(const uint4*)p;
    f.u[4] = t.x; f.u[5] = t.y; f.u[6] = t.z; f.u[7] = t.w;
}

// ---------------- embedding (x stays f32: residual stream) ----------------
__global__ __launch_bounds__(256) void embed_kernel(const int* __restrict__ idx,
                                                    const float* __restrict__ tok,
                                                    const float* __restrict__ pos,
                                                    float* __restrict__ x) {
    int bt = blockIdx.x;
    int t  = bt & (Tt - 1);
    int d  = threadIdx.x * 4;
    int token = idx[bt];
    float4 a = *(const float4*)(tok + (size_t)token * Dm + d);
    float4 p = *(const float4*)(pos + (size_t)t * Dm + d);
    float4 o;
    o.x = a.x + p.x; o.y = a.y + p.y; o.z = a.z + p.z; o.w = a.w + p.w;
    *(float4*)(x + (size_t)bt * Dm + d) = o;
}

// ---------------- LayerNorm: one wave per row; bf16 output ----------------
__global__ __launch_bounds__(256) void ln_kernel(const float* __restrict__ x,
                                                 const float* __restrict__ g,
                                                 const float* __restrict__ b,
                                                 unsigned* __restrict__ out) { // bf16 pairs
    int row  = blockIdx.x * 8 + (threadIdx.x >> 5);
    int lane = threadIdx.x & 31;
    const float* xr = x + (size_t)row * Dm;
    float4 vb[8];
    float s = 0.f, s2 = 0.f;
#pragma unroll
    for (int i = 0; i < 8; ++i) {
        float4 t = *(const float4*)(xr + i * 128 + lane * 4);
        vb[i] = t;
        s  += t.x + t.y + t.z + t.w;
        s2 += t.x * t.x + t.y * t.y + t.z * t.z + t.w * t.w;
    }
    s  = wave_sum(s);
    s2 = wave_sum(s2);
    float mean = s * (1.0f / Dm);
    float var  = s2 * (1.0f / Dm) - mean * mean;
    float rstd = rsqrtf(var + LN_EPS);
    unsigned* orow = out + (size_t)row * (Dm / 2);
#pragma unroll
    for (int i = 0; i < 8; ++i) {
        int d = i * 128 + lane * 4;
        float4 gv = *(const float4*)(g + d);
        float4 bv = *(const float4*)(b + d);
        float4 t = vb[i];
        float ox = (t.x - mean) * rstd * gv.x + bv.x;
        float oy = (t.y - mean) * rstd * gv.y + bv.y;
        float oz = (t.z - mean) * rstd * gv.z + bv.z;
        float ow = (t.w - mean) * rstd * gv.w + bv.w;
        orow[(d >> 1)]     = pack2(ox, oy);
        orow[(d >> 1) + 1] = pack2(oz, ow);
    }
}

// ---------------- generic WMMA GEMM ----------------
// C[M,N] = act(A_bf16[M,K] @ W_f32[K,N] + bias) (+ residual if f32 out).
// M%128==0, N%64==0, K%32==0. Block 256 (8 waves); tile 128x64; BK=32.
// A tile staged via GLOBAL_LOAD_ASYNC_TO_LDS_B128 (ASYNCcnt path).
#define AB_STRIDE 20   // uints per LDS row: 80B, 16B-aligned (b128 DS/async)
__global__ __launch_bounds__(256) void gemm_kernel(const unsigned short* __restrict__ A,
                                                   const float* __restrict__ W,
                                                   const float* __restrict__ bias,
                                                   const float* __restrict__ residual,
                                                   void* __restrict__ Cout,
                                                   int M, int N, int K,
                                                   int relu, int out_bf16) {
    __shared__ unsigned Asu[128 * AB_STRIDE];  // 128 rows x 16 bf16-pairs (+pad)
    __shared__ unsigned Bsu[64 * AB_STRIDE];   // 64 cols x 16 K-pairs (transposed)
    unsigned short* Bss = (unsigned short*)Bsu;

    int tid = threadIdx.x, wave = tid >> 5, lane = tid & 31, mn = lane & 15;
    int row0 = blockIdx.y * 128, col0 = blockIdx.x * 64;
    int kbA = (lane < 16) ? 0 : 8;
    int kbB = (lane < 16) ? 0 : 8;

    v8f zero = {0.f, 0.f, 0.f, 0.f, 0.f, 0.f, 0.f, 0.f};
    v8f acc[4] = {zero, zero, zero, zero};

    // A staging: 2 threads per row, 2 x 16B async copies each (loop-invariant LDS dst)
    int arow  = tid >> 1;
    int ahalf = tid & 1;
    unsigned ldsA0 = (unsigned)(size_t)(&Asu[arow * AB_STRIDE + ahalf * 8]);
    unsigned ldsA1 = ldsA0 + 16;
    const unsigned short* aptr = A + (size_t)(row0 + arow) * K + ahalf * 16;

    for (int k0 = 0; k0 < K; k0 += 32) {
        // stage A tile (128x32 bf16): async global -> LDS, no VGPR round-trip
        {
            const unsigned short* ap = aptr + k0;
            asm volatile(
                "global_load_async_to_lds_b128 %0, %2, off\n\t"
                "global_load_async_to_lds_b128 %1, %3, off"
                :: "v"(ldsA0), "v"(ldsA1), "v"(ap), "v"(ap + 8)
                : "memory");
        }
        // stage B tile (32x64 f32) transposed -> Bss[n][k], native cvt
#pragma unroll
        for (int i = 0; i < 2; ++i) {
            int f4 = tid + i * 256;
            int kk = f4 >> 4;
            int nc = (f4 & 15) * 4;
            float4 t = *(const float4*)(W + (size_t)(k0 + kk) * N + col0 + nc);
            Bss[(nc + 0) * (2 * AB_STRIDE) + kk] = f2bf(t.x);
            Bss[(nc + 1) * (2 * AB_STRIDE) + kk] = f2bf(t.y);
            Bss[(nc + 2) * (2 * AB_STRIDE) + kk] = f2bf(t.z);
            Bss[(nc + 3) * (2 * AB_STRIDE) + kk] = f2bf(t.w);
        }
        if (k0 + 32 < K)
            __builtin_prefetch(W + (size_t)(k0 + 32 + (tid >> 3)) * N + col0, 0, 1);
        asm volatile("s_wait_asynccnt 0x0" ::: "memory");
        __syncthreads();

        // A fragment: 16x32, rows m = wave*16 + (lane&15); 2x ds_load_b128
        Frag af;
        int mrow = wave * 16 + mn;
        frag_lo(af, &Asu[mrow * AB_STRIDE + (kbA >> 1)]);
        frag_hi(af, &Asu[mrow * AB_STRIDE + 8 + (kbA >> 1)]);
        // preload all 4 B fragments (2x ds_load_b128 each), then 4 WMMAs
        Frag bfr[4];
#pragma unroll
        for (int sub = 0; sub < 4; ++sub) {
            int n = sub * 16 + mn;
            frag_lo(bfr[sub], &Bsu[n * AB_STRIDE + (kbB >> 1)]);
            frag_hi(bfr[sub], &Bsu[n * AB_STRIDE + (kbB >> 1) + 8]);
        }
#pragma unroll
        for (int sub = 0; sub < 4; ++sub)
            acc[sub] = wmma_bf16(af, bfr[sub], acc[sub]);
        __syncthreads();
    }

    // epilogue
    int rbase = (lane < 16) ? 0 : 8;
#pragma unroll
    for (int sub = 0; sub < 4; ++sub) {
        int col = col0 + sub * 16 + mn;
        float bb = bias ? bias[col] : 0.0f;
#pragma unroll
        for (int r = 0; r < 8; ++r) {
            int row = row0 + wave * 16 + rbase + r;
            float val = acc[sub][r] + bb;
            if (residual) val += residual[(size_t)row * N + col];
            if (relu) val = fmaxf(val, 0.0f);
            if (out_bf16)
                ((unsigned short*)Cout)[(size_t)row * N + col] = f2bf(val);
            else
                ((float*)Cout)[(size_t)row * N + col] = val;
        }
    }
}

// ---------------- attention scores: wei = scale * Q @ K^T ----------------
// q,k are bf16 [B,T,D]; one wave per (bh, q-tile16, k-tile16), causal skip
__global__ __launch_bounds__(32) void attn_scores_kernel(const unsigned short* __restrict__ q,
                                                         const unsigned short* __restrict__ k,
                                                         float* __restrict__ wei) {
    int bh = blockIdx.x, qt = blockIdx.y, kt = blockIdx.z;
    if (kt > qt) return;                    // causal: wave-uniform skip
    int b = bh >> 4, h = bh & 15;
    int lane = threadIdx.x;
    int mn = lane & 15;
    int q0 = qt * 16, k0 = kt * 16;
    int kbA = (lane < 16) ? 0 : 8;
    int kbB = (lane < 16) ? 0 : 8;

    const unsigned short* qbase = q + (size_t)(b * Tt + q0 + mn) * Dm + h * HD;
    const unsigned short* kbase = k + (size_t)(b * Tt + k0 + mn) * Dm + h * HD;

    v8f acc = {0.f, 0.f, 0.f, 0.f, 0.f, 0.f, 0.f, 0.f};
#pragma unroll
    for (int half = 0; half < 2; ++half) {
        int koff = half * 32;
        Frag af, bf_;
        // A frag: runs {kbA..kbA+7} and {16+kbA..} -> 2x b128
        frag_lo(af, qbase + koff + kbA);
        frag_hi(af, qbase + koff + 16 + kbA);
        // B frag: runs {kbB..kbB+7} and {16+kbB..} -> 2x b128
        frag_lo(bf_, kbase + koff + kbB);
        frag_hi(bf_, kbase + koff + kbB + 16);
        acc = wmma_bf16(af, bf_, acc);
    }
    const float scale = 0.03125f;           // D^-0.5 = 1/32
    int rbase = (lane < 16) ? 0 : 8;
    float* wrow = wei + ((size_t)bh * Tt + q0) * Tt + k0;
#pragma unroll
    for (int r = 0; r < 8; ++r)
        wrow[(size_t)(rbase + r) * Tt + mn] = acc[r] * scale;
}

// ---------------- softmax: one wave per row; rewrite row in place as
// zero-padded bf16 probabilities (so P@V is a dense GEMM) ----------------
__global__ __launch_bounds__(256) void attn_softmax_kernel(float* __restrict__ wei) {
    int row  = blockIdx.x * 8 + (threadIdx.x >> 5);
    int lane = threadIdx.x & 31;
    int t = row & (Tt - 1);
    float* wr = wei + (size_t)row * Tt;
    float vals[32];
    float m = -1e30f;
#pragma unroll
    for (int c = 0; c < 32; ++c) {
        int s = c * 32 + lane;
        float x = (s <= t) ? wr[s] : -1e30f;
        vals[c] = x;
        m = fmaxf(m, x);
    }
    m = wave_max(m);
    float sum = 0.f;
#pragma unroll
    for (int c = 0; c < 32; ++c) {
        int s = c * 32 + lane;
        float e = (s <= t) ? expf(vals[c] - m) : 0.f;
        vals[c] = e;
        sum += e;
    }
    sum = wave_sum(sum);
    float inv = 1.0f / sum;
    unsigned short* wp = (unsigned short*)wr;   // reuse buffer: bf16 P, T entries
#pragma unroll
    for (int c = 0; c < 32; ++c) {
        int s = c * 32 + lane;
        wp[s] = (s <= t) ? f2bf(vals[c] * inv) : (unsigned short)0;
    }
}

// ---------------- att = P @ V (v bf16, att out bf16) ----------------
// one wave per (bh, q-tile16, n-subtile16)
__global__ __launch_bounds__(128) void attn_av_kernel(const unsigned short* __restrict__ P,
                                                      const unsigned short* __restrict__ v,
                                                      unsigned short* __restrict__ att) {
    int bh = blockIdx.x, qt = blockIdx.y;
    int b = bh >> 4, h = bh & 15;
    int tid = threadIdx.x, wave = tid >> 5, lane = tid & 31, mn = lane & 15;
    int q0 = qt * 16;
    int col = h * HD + wave * 16 + mn;
    int kbA = (lane < 16) ? 0 : 8;
    int kbB = (lane < 16) ? 0 : 16;
    const unsigned short* prow = P + (size_t)(bh * Tt + q0 + mn) * (2 * Tt);

    v8f acc = {0.f, 0.f, 0.f, 0.f, 0.f, 0.f, 0.f, 0.f};
    for (int ks = 0; ks < Tt / 32; ++ks) {
        int s0 = ks * 32;
        Frag af, bf_;
        // P frag: runs {s0+kbA..+7} and {s0+16+kbA..} -> 2x b128
        frag_lo(af, prow + s0 + kbA);
        frag_hi(af, prow + s0 + 16 + kbA);
#pragma unroll
        for (int i = 0; i < 8; ++i) {
            int s = s0 + kbA + 2 * i + (i >= 4 ? 8 : 0);
            unsigned a0 = v[(size_t)(b * Tt + s)     * Dm + col];
            unsigned a1 = v[(size_t)(b * Tt + s + 1) * Dm + col];
            bf_.u[i] = a0 | (a1 << 16);
        }
        acc = wmma_bf16(af, bf_, acc);
    }
    int rbase = (lane < 16) ? 0 : 8;
#pragma unroll
    for (int r = 0; r < 8; ++r)
        att[(size_t)(b * Tt + q0 + rbase + r) * Dm + col] = f2bf(acc[r]);
}

// ---------------- loss: per-row log-softmax gather ----------------
__global__ __launch_bounds__(256) void row_loss_kernel(const float* __restrict__ logits,
                                                       const int* __restrict__ y,
                                                       float* __restrict__ rowloss) {
    int row = blockIdx.x, tid = threadIdx.x;
    const float* lr = logits + (size_t)row * Vv;
    __shared__ float red[8];
    float m = -1e30f;
    for (int i = tid; i < Vv; i += 256) m = fmaxf(m, lr[i]);
    m = wave_max(m);
    if ((tid & 31) == 0) red[tid >> 5] = m;
    __syncthreads();
    float mm = red[0];
#pragma unroll
    for (int j = 1; j < 8; ++j) mm = fmaxf(mm, red[j]);
    __syncthreads();
    float s = 0.f;
    for (int i = tid; i < Vv; i += 256) s += expf(lr[i] - mm);
    s = wave_sum(s);
    if ((tid & 31) == 0) red[tid >> 5] = s;
    __syncthreads();
    if (tid == 0) {
        float tot = 0.f;
#pragma unroll
        for (int j = 0; j < 8; ++j) tot += red[j];
        int yi = y[row];
        rowloss[row] = -(lr[yi] - mm - logf(tot));
    }
}

__global__ __launch_bounds__(256) void loss_reduce_kernel(const float* __restrict__ rowloss,
                                                          float* __restrict__ out) {
    __shared__ float red[8];
    int tid = threadIdx.x;
    float s = 0.f;
    for (int i = tid; i < BT; i += 256) s += rowloss[i];
    s = wave_sum(s);
    if ((tid & 31) == 0) red[tid >> 5] = s;
    __syncthreads();
    if (tid == 0) {
        float tot = 0.f;
#pragma unroll
        for (int j = 0; j < 8; ++j) tot += red[j];
        out[0] = tot / (float)BT;
    }
}

// ---------------- host launch ----------------
extern "C" void kernel_launch(void* const* d_in, const int* in_sizes, int n_in,
                              void* d_out, int out_size, void* d_ws, size_t ws_size,
                              hipStream_t stream) {
    (void)in_sizes; (void)n_in; (void)out_size; (void)ws_size;
    const int*   idx   = (const int*)d_in[0];
    const int*   y     = (const int*)d_in[1];
    const float* tok   = (const float*)d_in[2];
    const float* pos   = (const float*)d_in[3];
    const float* Wq    = (const float*)d_in[4];
    const float* Wk    = (const float*)d_in[5];
    const float* Wv    = (const float*)d_in[6];
    const float* Wo    = (const float*)d_in[7];
    const float* bo    = (const float*)d_in[8];
    const float* ln1g  = (const float*)d_in[9];
    const float* ln1b  = (const float*)d_in[10];
    const float* ln2g  = (const float*)d_in[11];
    const float* ln2b  = (const float*)d_in[12];
    const float* W1    = (const float*)d_in[13];
    const float* b1    = (const float*)d_in[14];
    const float* W2    = (const float*)d_in[15];
    const float* b2    = (const float*)d_in[16];
    const float* lnfg  = (const float*)d_in[17];
    const float* lnfb  = (const float*)d_in[18];
    const float* lmw   = (const float*)d_in[19];
    const float* lmb   = (const float*)d_in[20];

    float* logits = (float*)d_out;                      // [BT, V]
    float* loss   = logits + (size_t)BT * Vv;           // scalar

    // workspace layout (bytes)
    char* p = (char*)d_ws;
    float* x = (float*)p;               p += (size_t)BT * Dm * 4;      // f32 residual
    unsigned short* h   = (unsigned short*)p; p += (size_t)BT * Dm * 2; // bf16
    unsigned short* q   = (unsigned short*)p; p += (size_t)BT * Dm * 2;
    unsigned short* k   = (unsigned short*)p; p += (size_t)BT * Dm * 2;
    unsigned short* v   = (unsigned short*)p; p += (size_t)BT * Dm * 2;
    unsigned short* att = (unsigned short*)p; p += (size_t)BT * Dm * 2;
    unsigned short* ff  = (unsigned short*)p; p += (size_t)BT * 4 * Dm * 2;
    float* wei  = (float*)p;            p += (size_t)Bb * Hh * Tt * Tt * 4;
    float* rowl = (float*)p;

    dim3 blk256(256), blk128(128), blk32(32);
    dim3 gLN(BT / 8);
    dim3 gGemmD(Dm / 64, BT / 128);                     // (16,16)
    dim3 gGemm4D(4 * Dm / 64, BT / 128);                // (64,16)
    dim3 gGemmV(Vv / 64, BT / 128);                     // (500,16)
    dim3 gScore(Bb * Hh, Tt / 16, Tt / 16);             // (32,64,64)
    dim3 gSoft(Bb * Hh * Tt / 8);                       // 4096
    dim3 gAV(Bb * Hh, Tt / 16);                         // (32,64)

    embed_kernel<<<BT, blk256, 0, stream>>>(idx, tok, pos, x);

    for (int l = 0; l < Ll; ++l) {
        const size_t wO = (size_t)l * Dm * Dm;
        const size_t w1 = (size_t)l * Dm * 4 * Dm;
        const size_t bD = (size_t)l * Dm;
        const size_t b4 = (size_t)l * 4 * Dm;

        ln_kernel<<<gLN, blk256, 0, stream>>>(x, ln1g + bD, ln1b + bD, (unsigned*)h);
        gemm_kernel<<<gGemmD, blk256, 0, stream>>>(h, Wq + wO, nullptr, nullptr, q, BT, Dm, Dm, 0, 1);
        gemm_kernel<<<gGemmD, blk256, 0, stream>>>(h, Wk + wO, nullptr, nullptr, k, BT, Dm, Dm, 0, 1);
        gemm_kernel<<<gGemmD, blk256, 0, stream>>>(h, Wv + wO, nullptr, nullptr, v, BT, Dm, Dm, 0, 1);
        attn_scores_kernel<<<gScore, blk32, 0, stream>>>(q, k, wei);
        attn_softmax_kernel<<<gSoft, blk256, 0, stream>>>(wei);
        attn_av_kernel<<<gAV, blk128, 0, stream>>>((const unsigned short*)wei, v, att);
        gemm_kernel<<<gGemmD, blk256, 0, stream>>>(att, Wo + wO, bo + bD, x, x, BT, Dm, Dm, 0, 0);
        ln_kernel<<<gLN, blk256, 0, stream>>>(x, ln2g + bD, ln2b + bD, (unsigned*)h);
        gemm_kernel<<<gGemm4D, blk256, 0, stream>>>(h, W1 + w1, b1 + b4, nullptr, ff, BT, 4 * Dm, Dm, 1, 1);
        gemm_kernel<<<gGemmD, blk256, 0, stream>>>(ff, W2 + w1, b2 + bD, x, x, BT, Dm, 4 * Dm, 0, 0);
    }

    ln_kernel<<<gLN, blk256, 0, stream>>>(x, lnfg, lnfb, (unsigned*)h);
    gemm_kernel<<<gGemmV, blk256, 0, stream>>>(h, lmw, lmb, nullptr, logits, BT, Vv, Dm, 0, 0);
    row_loss_kernel<<<BT, blk256, 0, stream>>>(logits, y, rowl);
    loss_reduce_kernel<<<1, blk256, 0, stream>>>(rowl, loss);
}
